// GraphRec_26027501813838
// MI455X (gfx1250) — matrix operand order repaired and checked by reading, not compiled, not run on hardware
//
#include <hip/hip_runtime.h>

// ---------------------------------------------------------------------------
// GraphRec forward for MI455X (gfx1250, wave32, WMMA).
// Heavy GEMMs use v_wmma_f32_16x16x32_f16 (f16 in / f32 accumulate).
// Social-branch softmax is over the BATCH axis -> two-pass with recompute.
// ---------------------------------------------------------------------------

constexpr int kD = 64;     // embedding dim
constexpr int kN = 30;     // neighbors
constexpr int kB = 512;    // batch

typedef __attribute__((ext_vector_type(16))) _Float16 v16h;
typedef __attribute__((ext_vector_type(8)))  float    v8f;

// ---- WMMA fragment maps (this code's contiguous k order, same for A and B) --------------------------------
// A (16x32 f16): lane<16 -> K = j; lane>=16 -> K = 16 + j
__device__ __forceinline__ int amap(int j, int lane) {
  return j + ((lane & 16) ? 16 : 0);
}
// B (32x16 f16): lanes 0-15 hold K=0..15 (half j -> K=j), lanes 16-31 K=16..31
__device__ __forceinline__ int bmap(int j, int lane) {
  return j + ((lane >> 4) & 1) * 16;
}

__device__ __forceinline__ v16h ldA(const _Float16* Xs, int ld, int lane) {
  v16h a;
  const _Float16* row = Xs + (lane & 15) * ld;
#pragma unroll
  for (int j = 0; j < 16; ++j) a[j] = row[amap(j, lane)];
  return a;
}

// packed B fragments: [frag][lane][16 halves] contiguous -> 32B/lane load
__device__ __forceinline__ v16h ldB(const _Float16* Wp, int frag, int lane) {
  return ((const v16h*)(Wp + (size_t)frag * 512))[lane];
}

// out = opt_relu( X(16 x 32*ktiles) @ W(.. x 64) + bias ), C layout:
// lane l, vgpr r -> (m = r + 8*(l>>4), n = nt*16 + (l&15))
__device__ __forceinline__ void gemm16(const _Float16* Xs, int ldx, int ktiles,
                                       const _Float16* Wp, const float* bias,
                                       bool do_relu, _Float16* Out, int ldo,
                                       v8f* Cout, int lane) {
  const int n0 = lane & 15;
  const int hi = (lane >> 4) & 1;
#pragma unroll
  for (int nt = 0; nt < 4; ++nt) {
    v8f c = {};
    for (int kt = 0; kt < ktiles; ++kt) {
      v16h a = ldA(Xs + kt * 32, ldx, lane);
      v16h b = ldB(Wp, kt * 4 + nt, lane);
      c = __builtin_amdgcn_wmma_f32_16x16x32_f16(false, a, false, b,
                                                 (short)0, c, false, false);
    }
    const float bn = bias[nt * 16 + n0];
#pragma unroll
    for (int r = 0; r < 8; ++r) {
      float v = c[r] + bn;
      if (do_relu) v = fmaxf(v, 0.f);
      c[r] = v;
      if (Out) Out[(r + 8 * hi) * ldo + nt * 16 + n0] = (_Float16)v;
    }
    if (Cout) Cout[nt] = c;
  }
}

__device__ __forceinline__ void cp64(const float* src, _Float16* dst) {
#pragma unroll
  for (int k = 0; k < 64; k += 4) {
    float4 v = *(const float4*)(src + k);
    dst[k] = (_Float16)v.x; dst[k + 1] = (_Float16)v.y;
    dst[k + 2] = (_Float16)v.z; dst[k + 3] = (_Float16)v.w;
  }
}

// MLP3 on one 16-row tile staged in LDS (Xs 16x128 -> Hs -> Gs -> X frags).
// If x16, also writes the f32->f16 result into Xs[:,64:128] for attention.
__device__ __forceinline__ void mlp3(_Float16* Xs, _Float16* Hs, _Float16* Gs,
                                     const _Float16* W1, const float* b1,
                                     const _Float16* W2, const float* b2,
                                     const _Float16* W3, const float* b3,
                                     v8f X[4], bool x16, int lane) {
  gemm16(Xs, 128, 4, W1, b1, true, Hs, 64, nullptr, lane);
  __builtin_amdgcn_wave_barrier();
  gemm16(Hs, 64, 2, W2, b2, true, Gs, 64, nullptr, lane);
  __builtin_amdgcn_wave_barrier();
  gemm16(Gs, 64, 2, W3, b3, false, x16 ? (Xs + 64) : (_Float16*)nullptr, 128,
         X, lane);
  __builtin_amdgcn_wave_barrier();
}

// ---------------------------------------------------------------------------
// Weight fragment packing (one permutation pass, f32 -> f16)
// ---------------------------------------------------------------------------
struct ConvArgs {
  const float* src[16];
  int dstOff[16];  // in halves
  int kt[16];
};

__global__ void k_conv(ConvArgs A, _Float16* base) {
  int m = blockIdx.y;
  int e = blockIdx.x * 256 + threadIdx.x;
  if (e >= A.kt[m] * 2048) return;
  int frag = e >> 9;
  int lane = (e >> 4) & 31;
  int j = e & 15;
  int kt = frag >> 2, nt = frag & 3;
  int K = kt * 32 + bmap(j, lane);
  int n = nt * 16 + (lane & 15);
  base[A.dstOff[m] + e] = (_Float16)A.src[m][K * 64 + n];
}

// ---------------------------------------------------------------------------
// Main fused kernel: gather -> MLP3 -> attention score
// mode 0: xia  (items of user)      mode 1: social pass1 (scores only)
// mode 2: fjt  (users of item)      mode 3: plain concat-MLP3 (umlp)
// mode 4: attention-only (attS over precomputed ho_I)
// ---------------------------------------------------------------------------
struct MainArgs {
  int mode, rows;
  const float *tabA, *tabR, *tabRep;
  const int *idxA, *idxR, *idxRep;
  const float *dirL, *dirR;
  const _Float16 *W1, *W2, *W3;
  const float *b1, *b2, *b3;
  const _Float16* aW1;
  const float *ab1, *aW2, *ab2;
  float* xout;
  float* sout;
};

__global__ __launch_bounds__(128) void k_main(MainArgs A) {
  __shared__ _Float16 lds[4][4096];
  const int wave = threadIdx.x >> 5, lane = threadIdx.x & 31;
  const int tile = blockIdx.x * 4 + wave;
  const int row0 = tile * 16;
  if (row0 >= A.rows) return;
  _Float16* Xs = lds[wave];
  _Float16* Hs = Xs + 2048;
  _Float16* Gs = Hs + 1024;
  const int lr = lane >> 1, lh = lane & 1;
  const int r = row0 + lr;

  if (A.mode <= 2) {
    const float* src = lh ? (A.tabR + (size_t)A.idxR[r] * 64)
                          : (A.tabA + (size_t)A.idxA[r] * 64);
    cp64(src, Xs + lr * 128 + lh * 64);
  } else if (A.mode == 3) {
    cp64((lh ? A.dirR : A.dirL) + (size_t)r * 64, Xs + lr * 128 + lh * 64);
  } else {  // mode 4: concat(rep, neighbor)
    const float* src = lh ? (A.dirR + (size_t)r * 64)
                          : (A.tabRep + (size_t)A.idxRep[r / kN] * 64);
    cp64(src, Xs + lr * 128 + lh * 64);
  }
  __builtin_amdgcn_wave_barrier();

  if (A.mode != 4) {
    v8f X[4];
    mlp3(Xs, Hs, Gs, A.W1, A.b1, A.W2, A.b2, A.W3, A.b3, X, A.mode <= 2, lane);
    if (A.xout) {
      const int hi = (lane >> 4) & 1, n0 = lane & 15;
#pragma unroll
      for (int nt = 0; nt < 4; ++nt)
#pragma unroll
        for (int rr = 0; rr < 8; ++rr)
          A.xout[(size_t)(row0 + rr + 8 * hi) * 64 + nt * 16 + n0] = X[nt][rr];
    }
  }
  if (A.sout) {
    if (A.mode <= 2) {  // stage self-rep into left half of Xs
      if (lh == 0) {
        int repi;
        if (A.mode == 1) {
          int b = r / (kN * kN), j = r % kN;
          repi = A.idxRep[b * kN + j];  // rep[b,i,j] = po[b,j]
        } else {
          repi = A.idxRep[r / kN];
        }
        cp64(A.tabRep + (size_t)repi * 64, Xs + lr * 128);
      }
      __builtin_amdgcn_wave_barrier();
    }
    gemm16(Xs, 128, 4, A.aW1, A.ab1, true, Hs, 64, nullptr, lane);
    __builtin_amdgcn_wave_barrier();
    if (lane < 16) {
      float s = A.ab2[0];
      for (int k = 0; k < 64; ++k) s += (float)Hs[lane * 64 + k] * A.aW2[k];
      A.sout[row0 + lane] = s;
    }
  }
}

// ---------------------------------------------------------------------------
// Softmax stats over the batch axis: for column c, max/denominator over b
// ---------------------------------------------------------------------------
__global__ void k_stats(const float* S, int M, float* mx, float* den) {
  const int c = blockIdx.x, t = threadIdx.x;  // 64 threads
  __shared__ float sm[64];
  float m = -1e30f;
  for (int b = t; b < kB; b += 64) m = fmaxf(m, S[(size_t)b * M + c]);
  sm[t] = m; __syncthreads();
  for (int s = 32; s > 0; s >>= 1) {
    if (t < s) sm[t] = fmaxf(sm[t], sm[t + s]);
    __syncthreads();
  }
  float mv = sm[0]; __syncthreads();
  float d = 0.f;
  for (int b = t; b < kB; b += 64) d += expf(S[(size_t)b * M + c] - mv);
  sm[t] = d; __syncthreads();
  for (int s = 32; s > 0; s >>= 1) {
    if (t < s) sm[t] += sm[t + s];
    __syncthreads();
  }
  if (t == 0) { mx[c] = mv; den[c] = sm[0]; }
}

// out[b,:] = sum_n softmax_b(S)[b,n] * X[b,n,:]
__global__ void k_wsum(const float* S, const float* mx, const float* den,
                       const float* X, float* out) {
  const int b = blockIdx.x, k = threadIdx.x;  // 64 threads
  float acc = 0.f;
  for (int n = 0; n < kN; ++n) {
    float w = expf(S[(size_t)b * kN + n] - mx[n]) / den[n];
    acc += w * X[((size_t)b * kN + n) * 64 + k];
  }
  out[(size_t)b * 64 + k] = acc;
}

// ---------------------------------------------------------------------------
// Social pass 2: per (b,j), recompute xoa rows (i=0..29) via WMMA MLP3 and
// accumulate ho_pre[b,j,:] = sum_i alpha[b,i,j] * xoa[b,i,j,:]
// ---------------------------------------------------------------------------
struct SocArgs {
  const float *E_i, *E_r;
  const int *uui, *uuir;
  const _Float16 *W1, *W2, *W3;
  const float *b1, *b2, *b3;
  const float *s, *mx, *den;
  float* out;
};

__global__ __launch_bounds__(128) void k_social(SocArgs A) {
  __shared__ _Float16 lds[4][4096];
  const int wave = threadIdx.x >> 5, lane = threadIdx.x & 31;
  const int task = blockIdx.x * 4 + wave;  // task = b*kN + j
  if (task >= kB * kN) return;
  const int b = task / kN, j = task % kN;
  _Float16* Xs = lds[wave];
  _Float16* Hs = Xs + 2048;
  _Float16* Gs = Hs + 1024;
  const int lr = lane >> 1, lh = lane & 1;
  const int hi = (lane >> 4) & 1;
  float acc0 = 0.f, acc1 = 0.f, acc2 = 0.f, acc3 = 0.f;
  for (int t = 0; t < 2; ++t) {
    const int i = t * 16 + lr;
    _Float16* dst = Xs + lr * 128 + lh * 64;
    if (i < kN) {
      size_t e = ((size_t)(b * kN + i)) * kN + j;
      const float* src = lh ? (A.E_r + (size_t)A.uuir[e] * 64)
                            : (A.E_i + (size_t)A.uui[e] * 64);
      cp64(src, dst);
    } else {
      for (int k = 0; k < 64; ++k) dst[k] = (_Float16)0.f;
    }
    __builtin_amdgcn_wave_barrier();
    v8f X[4];
    mlp3(Xs, Hs, Gs, A.W1, A.b1, A.W2, A.b2, A.W3, A.b3, X, false, lane);
    float al[8];
#pragma unroll
    for (int rr = 0; rr < 8; ++rr) {
      const int ii = t * 16 + rr + 8 * hi;
      float a = 0.f;
      if (ii < kN) {
        const int col = ii * kN + j;
        a = expf(A.s[(size_t)b * (kN * kN) + col] - A.mx[col]) / A.den[col];
      }
      al[rr] = a;
    }
#pragma unroll
    for (int rr = 0; rr < 8; ++rr) {
      acc0 += al[rr] * X[0][rr];
      acc1 += al[rr] * X[1][rr];
      acc2 += al[rr] * X[2][rr];
      acc3 += al[rr] * X[3][rr];
    }
    __builtin_amdgcn_wave_barrier();
  }
  acc0 += __shfl_xor(acc0, 16, 32);
  acc1 += __shfl_xor(acc1, 16, 32);
  acc2 += __shfl_xor(acc2, 16, 32);
  acc3 += __shfl_xor(acc3, 16, 32);
  if (lane < 16) {
    float* o = A.out + (size_t)task * 64 + lane;
    o[0] = acc0; o[16] = acc1; o[32] = acc2; o[48] = acc3;
  }
}

// ---------------------------------------------------------------------------
// out = opt_relu(in @ W(64x64) + b) over `rows` rows (WMMA)
// ---------------------------------------------------------------------------
struct LinArgs {
  int rows, relu;
  const float* in;
  const _Float16* W;
  const float* b;
  float* out;
};

__global__ __launch_bounds__(128) void k_linear(LinArgs A) {
  __shared__ _Float16 lds[4][1024];
  const int wave = threadIdx.x >> 5, lane = threadIdx.x & 31;
  const int tile = blockIdx.x * 4 + wave;
  const int row0 = tile * 16;
  if (row0 >= A.rows) return;
  _Float16* Xs = lds[wave];
  const int lr = lane >> 1, lh = lane & 1;
  {
    const float* src = A.in + (size_t)(row0 + lr) * 64 + lh * 32;
    _Float16* dst = Xs + lr * 64 + lh * 32;
#pragma unroll
    for (int k = 0; k < 32; k += 4) {
      float4 v = *(const float4*)(src + k);
      dst[k] = (_Float16)v.x; dst[k + 1] = (_Float16)v.y;
      dst[k + 2] = (_Float16)v.z; dst[k + 3] = (_Float16)v.w;
    }
  }
  __builtin_amdgcn_wave_barrier();
  v8f C[4];
  gemm16(Xs, 64, 2, A.W, A.b, A.relu != 0, nullptr, 0, C, lane);
  const int hi = (lane >> 4) & 1, n0 = lane & 15;
#pragma unroll
  for (int nt = 0; nt < 4; ++nt)
#pragma unroll
    for (int r = 0; r < 8; ++r)
      A.out[(size_t)(row0 + r + 8 * hi) * 64 + nt * 16 + n0] = C[nt][r];
}

// final g-MLP: [B,128] @ [128,1] -> relu -> 1x1 -> relu -> 1x1
__global__ void k_final(const float* hi, const float* zj, const float* gW1,
                        const float* gb1, const float* gW2, const float* gb2,
                        const float* gW3, const float* gb3, float* out) {
  const int b = blockIdx.x * 64 + threadIdx.x;
  if (b >= kB) return;
  float s = gb1[0];
  for (int k = 0; k < 64; ++k) s += hi[(size_t)b * 64 + k] * gW1[k];
  for (int k = 0; k < 64; ++k) s += zj[(size_t)b * 64 + k] * gW1[64 + k];
  s = fmaxf(s, 0.f);
  s = fmaxf(s * gW2[0] + gb2[0], 0.f);
  out[b] = s * gW3[0] + gb3[0];
}

// ---------------------------------------------------------------------------
extern "C" void kernel_launch(void* const* d_in, const int* in_sizes, int n_in,
                              void* d_out, int out_size, void* d_ws,
                              size_t ws_size, hipStream_t stream) {
  (void)in_sizes; (void)n_in; (void)out_size; (void)ws_size;
  const float* E_u = (const float*)d_in[0];
  const float* E_i = (const float*)d_in[1];
  const float* E_r = (const float*)d_in[2];
  const float *gv_W1 = (const float*)d_in[3],  *gv_b1 = (const float*)d_in[4];
  const float *gv_W2 = (const float*)d_in[5],  *gv_b2 = (const float*)d_in[6];
  const float *gv_W3 = (const float*)d_in[7],  *gv_b3 = (const float*)d_in[8];
  const float *umlp_W1 = (const float*)d_in[9],  *umlp_b1 = (const float*)d_in[10];
  const float *umlp_W2 = (const float*)d_in[11], *umlp_b2 = (const float*)d_in[12];
  const float *umlp_W3 = (const float*)d_in[13], *umlp_b3 = (const float*)d_in[14];
  const float *gu_W1 = (const float*)d_in[15], *gu_b1 = (const float*)d_in[16];
  const float *gu_W2 = (const float*)d_in[17], *gu_b2 = (const float*)d_in[18];
  const float *gu_W3 = (const float*)d_in[19], *gu_b3 = (const float*)d_in[20];
  const float *attI_W1 = (const float*)d_in[21], *attI_b1 = (const float*)d_in[22];
  const float *attI_W2 = (const float*)d_in[23], *attI_b2 = (const float*)d_in[24];
  const float *attO_W1 = (const float*)d_in[25], *attO_b1 = (const float*)d_in[26];
  const float *attO_W2 = (const float*)d_in[27], *attO_b2 = (const float*)d_in[28];
  const float *attS_W1 = (const float*)d_in[29], *attS_b1 = (const float*)d_in[30];
  const float *attS_W2 = (const float*)d_in[31], *attS_b2 = (const float*)d_in[32];
  const float *atti_W1 = (const float*)d_in[33], *atti_b1 = (const float*)d_in[34];
  const float *atti_W2 = (const float*)d_in[35], *atti_b2 = (const float*)d_in[36];
  const float *linI_W = (const float*)d_in[37], *linI_b = (const float*)d_in[38];
  const float *linO_W = (const float*)d_in[39], *linO_b = (const float*)d_in[40];
  const float *ilin_W = (const float*)d_in[41], *ilin_b = (const float*)d_in[42];
  const float *g_W1 = (const float*)d_in[43], *g_b1 = (const float*)d_in[44];
  const float *g_W2 = (const float*)d_in[45], *g_b2 = (const float*)d_in[46];
  const float *g_W3 = (const float*)d_in[47], *g_b3 = (const float*)d_in[48];
  const int* nodes_u = (const int*)d_in[49];
  const int* nodes_i = (const int*)d_in[50];
  const int* u_items_list = (const int*)d_in[51];
  const int* u_users_list = (const int*)d_in[52];
  const int* i_users_list = (const int*)d_in[53];
  const int* u_items_r = (const int*)d_in[54];
  const int* u_users_items = (const int*)d_in[55];
  const int* u_users_items_r = (const int*)d_in[56];
  const int* i_users_r = (const int*)d_in[57];

  // ---- workspace bump allocator (256B aligned) ----
  char* ws = (char*)d_ws;
  size_t off = 0;
  auto alloc = [&](size_t bytes) -> char* {
    char* p = ws + off;
    off = (off + bytes + 255) & ~(size_t)255;
    return p;
  };

  // packed f16 weight fragments
  const float* srcs[16] = {gv_W1, gv_W2, gv_W3, gu_W1, gu_W2, gu_W3,
                           umlp_W1, umlp_W2, umlp_W3, attI_W1, attO_W1,
                           attS_W1, atti_W1, linI_W, linO_W, ilin_W};
  const int kts[16] = {4, 2, 2, 4, 2, 2, 4, 2, 2, 4, 4, 4, 4, 2, 2, 2};
  ConvArgs ca;
  int offs[16], tot = 0;
  for (int m = 0; m < 16; ++m) {
    offs[m] = tot;
    tot += kts[m] * 2048;
    ca.src[m] = srcs[m];
    ca.dstOff[m] = offs[m];
    ca.kt[m] = kts[m];
  }
  _Float16* wf = (_Float16*)alloc((size_t)tot * 2);
  const _Float16 *f_gvW1 = wf + offs[0], *f_gvW2 = wf + offs[1],
                 *f_gvW3 = wf + offs[2], *f_guW1 = wf + offs[3],
                 *f_guW2 = wf + offs[4], *f_guW3 = wf + offs[5],
                 *f_umW1 = wf + offs[6], *f_umW2 = wf + offs[7],
                 *f_umW3 = wf + offs[8], *f_attI = wf + offs[9],
                 *f_attO = wf + offs[10], *f_attS = wf + offs[11],
                 *f_atti = wf + offs[12], *f_linI = wf + offs[13],
                 *f_linO = wf + offs[14], *f_ilin = wf + offs[15];

  float* xia = (float*)alloc((size_t)kB * kN * 64 * 4);
  float* fjt = (float*)alloc((size_t)kB * kN * 64 * 4);
  float* ho_pre = (float*)alloc((size_t)kB * kN * 64 * 4);
  float* ho_I = (float*)alloc((size_t)kB * kN * 64 * 4);
  float* s_ia = (float*)alloc((size_t)kB * kN * 4);
  float* s_i = (float*)alloc((size_t)kB * kN * 4);
  float* s_S = (float*)alloc((size_t)kB * kN * 4);
  float* s_oa = (float*)alloc((size_t)kB * kN * kN * 4);
  float* mx_ia = (float*)alloc(kN * 4);
  float* den_ia = (float*)alloc(kN * 4);
  float* mx_i = (float*)alloc(kN * 4);
  float* den_i = (float*)alloc(kN * 4);
  float* mx_S = (float*)alloc(kN * 4);
  float* den_S = (float*)alloc(kN * 4);
  float* mx_oa = (float*)alloc(kN * kN * 4);
  float* den_oa = (float*)alloc(kN * kN * 4);
  float* hiI_pre = (float*)alloc((size_t)kB * 64 * 4);
  float* hi_I = (float*)alloc((size_t)kB * 64 * 4);
  float* hiS_pre = (float*)alloc((size_t)kB * 64 * 4);
  float* hi_S = (float*)alloc((size_t)kB * 64 * 4);
  float* hi = (float*)alloc((size_t)kB * 64 * 4);
  float* zj_pre = (float*)alloc((size_t)kB * 64 * 4);
  float* zj = (float*)alloc((size_t)kB * 64 * 4);

  auto grid16 = [](int rows) { return (rows / 16 + 3) / 4; };

  // 0) pack weights into B-fragment layout
  k_conv<<<dim3(32, 16, 1), 256, 0, stream>>>(ca, wf);

  // 1) item-space: xia + attI scores
  MainArgs a0{};
  a0.mode = 0; a0.rows = kB * kN;
  a0.tabA = E_i; a0.tabR = E_r; a0.tabRep = E_u;
  a0.idxA = u_items_list; a0.idxR = u_items_r; a0.idxRep = nodes_u;
  a0.W1 = f_gvW1; a0.W2 = f_gvW2; a0.W3 = f_gvW3;
  a0.b1 = gv_b1; a0.b2 = gv_b2; a0.b3 = gv_b3;
  a0.aW1 = f_attI; a0.ab1 = attI_b1; a0.aW2 = attI_W2; a0.ab2 = attI_b2;
  a0.xout = xia; a0.sout = s_ia;
  k_main<<<grid16(kB * kN), 128, 0, stream>>>(a0);

  // 2) item-model: fjt + atti scores
  MainArgs a2 = a0;
  a2.mode = 2;
  a2.tabA = E_u; a2.tabRep = E_i;
  a2.idxA = i_users_list; a2.idxR = i_users_r; a2.idxRep = nodes_i;
  a2.W1 = f_guW1; a2.W2 = f_guW2; a2.W3 = f_guW3;
  a2.b1 = gu_b1; a2.b2 = gu_b2; a2.b3 = gu_b3;
  a2.aW1 = f_atti; a2.ab1 = atti_b1; a2.aW2 = atti_W2; a2.ab2 = atti_b2;
  a2.xout = fjt; a2.sout = s_i;
  k_main<<<grid16(kB * kN), 128, 0, stream>>>(a2);

  // 3) social pass 1: attO scores only (xoa recomputed in pass 2)
  MainArgs a1 = a0;
  a1.mode = 1; a1.rows = kB * kN * kN;
  a1.idxA = u_users_items; a1.idxR = u_users_items_r; a1.idxRep = u_users_list;
  a1.aW1 = f_attO; a1.ab1 = attO_b1; a1.aW2 = attO_W2; a1.ab2 = attO_b2;
  a1.xout = nullptr; a1.sout = s_oa;
  k_main<<<grid16(kB * kN * kN), 128, 0, stream>>>(a1);

  // 4) batch-axis softmax stats
  k_stats<<<kN, 64, 0, stream>>>(s_ia, kN, mx_ia, den_ia);
  k_stats<<<kN, 64, 0, stream>>>(s_i, kN, mx_i, den_i);
  k_stats<<<kN * kN, 64, 0, stream>>>(s_oa, kN * kN, mx_oa, den_oa);

  // 5) weighted sums (item space / item model)
  k_wsum<<<kB, 64, 0, stream>>>(s_ia, mx_ia, den_ia, xia, hiI_pre);
  k_wsum<<<kB, 64, 0, stream>>>(s_i, mx_i, den_i, fjt, zj_pre);

  // 6) social pass 2: recompute xoa, accumulate ho_pre
  SocArgs sa{E_i, E_r, u_users_items, u_users_items_r,
             f_gvW1, f_gvW2, f_gvW3, gv_b1, gv_b2, gv_b3,
             s_oa, mx_oa, den_oa, ho_pre};
  k_social<<<(kB * kN + 3) / 4, 128, 0, stream>>>(sa);

  // 7) linears
  LinArgs l0{kB, 1, hiI_pre, f_linI, linI_b, hi_I};
  k_linear<<<grid16(kB), 128, 0, stream>>>(l0);
  LinArgs l1{kB * kN, 1, ho_pre, f_linO, linO_b, ho_I};
  k_linear<<<grid16(kB * kN), 128, 0, stream>>>(l1);

  // 8) attS scores over ho_I
  MainArgs a4{};
  a4.mode = 4; a4.rows = kB * kN;
  a4.tabRep = E_u; a4.idxRep = nodes_u; a4.dirR = ho_I;
  a4.aW1 = f_attS; a4.ab1 = attS_b1; a4.aW2 = attS_W2; a4.ab2 = attS_b2;
  a4.sout = s_S;
  k_main<<<grid16(kB * kN), 128, 0, stream>>>(a4);
  k_stats<<<kN, 64, 0, stream>>>(s_S, kN, mx_S, den_S);
  k_wsum<<<kB, 64, 0, stream>>>(s_S, mx_S, den_S, ho_I, hiS_pre);
  LinArgs l2{kB, 1, hiS_pre, f_linO, linO_b, hi_S};  // reference reuses linO
  k_linear<<<grid16(kB), 128, 0, stream>>>(l2);

  // 9) hi = umlp(concat(hi_I, hi_S))
  MainArgs a3{};
  a3.mode = 3; a3.rows = kB;
  a3.dirL = hi_I; a3.dirR = hi_S;
  a3.W1 = f_umW1; a3.W2 = f_umW2; a3.W3 = f_umW3;
  a3.b1 = umlp_b1; a3.b2 = umlp_b2; a3.b3 = umlp_b3;
  a3.xout = hi; a3.sout = nullptr;
  k_main<<<grid16(kB), 128, 0, stream>>>(a3);

  // 10) zj
  LinArgs l3{kB, 1, zj_pre, f_ilin, ilin_b, zj};
  k_linear<<<grid16(kB), 128, 0, stream>>>(l3);

  // 11) rating head
  k_final<<<kB / 64, 64, 0, stream>>>(hi, zj, g_W1, g_b1, g_W2, g_b2, g_W3,
                                      g_b3, (float*)d_out);
}
